// CFTAOBlock2D_23708219474536
// MI455X (gfx1250) — compile-verified, matrix operand order checked by reading, not run on hardware
//
#include <hip/hip_runtime.h>
#include <math.h>

typedef _Float16 h16;
typedef __attribute__((ext_vector_type(8)))  _Float16 v8h;
typedef __attribute__((ext_vector_type(16))) _Float16 v16h;
typedef __attribute__((ext_vector_type(8)))  float    v8f;

#define WMMA(a,b,c) __builtin_amdgcn_wmma_f32_16x16x32_f16(false,(a),false,(b),(short)0,(c),false,false)

__device__ __forceinline__ v8f vzero8() {
  v8f z = {0.f,0.f,0.f,0.f,0.f,0.f,0.f,0.f};
  return z;
}

// ---- fragment element maps -------------------------------------------------
// 16-bit WMMA lane layout: half = (krem>>3)&1 ; i = (krem&7) | ((krem>>4)<<3)
// B-tile set (32k x 16n), NT n-tiles per k-tile: packed elem offset
__device__ __forceinline__ int packB(int k, int n, int NT) {
  int kt = k >> 5, krem = k & 31;
  int lane = (n & 15) | (((krem >> 3) & 1) << 4);
  int i = (krem & 7) | ((krem >> 4) << 3);
  return (((kt * NT + (n >> 4)) << 5) + lane) * 16 + i;
}
// A-tile set (16m x 32k), KT k-tiles per m-tile
__device__ __forceinline__ int packA(int m, int k, int KT) {
  int mt = m >> 4, kt = k >> 5, krem = k & 31;
  int lane = (m & 15) | (((krem >> 3) & 1) << 4);
  int i = (krem & 7) | ((krem >> 4) << 3);
  return ((((mt * KT + kt) << 5) + lane) << 4) + i;
}

// packed fragment: lane's 16 f16 contiguous -> two b128 LDS loads
__device__ __forceinline__ v16h frag_P(const h16* base, int tile, int lane) {
  const v8h* p = (const v8h*)(base + (((tile << 5) + lane) << 4));
  v8h lo = p[0], hi = p[1];
  return __builtin_shufflevector(lo, hi, 0,1,2,3,4,5,6,7,8,9,10,11,12,13,14,15);
}

// row-major A fragment (k contiguous): two b128 LDS loads
__device__ __forceinline__ v16h frag_A(const h16* p, int ld, int lane) {
  const h16* row = p + (lane & 15) * ld + ((lane >> 4) << 3);
  v8h lo = *(const v8h*)row;
  v8h hi = *(const v8h*)(row + 16);
  return __builtin_shufflevector(lo, hi, 0,1,2,3,4,5,6,7,8,9,10,11,12,13,14,15);
}

// ---- CDNA5 async global->LDS copy (ASYNCcnt path) --------------------------
__device__ __forceinline__ void async_issue_f16(h16* dst, const h16* src, int n_elems,
                                                int tid, int nthr) {
  int n16 = n_elems >> 3;                       // 16-byte granules
  for (int i = tid; i < n16; i += nthr) {
    unsigned loff = (unsigned)(uintptr_t)(dst + (i << 3));
    const void* g = (const void*)(src + (i << 3));
    asm volatile("global_load_async_to_lds_b128 %0, %1, off" :: "v"(loff), "v"(g) : "memory");
  }
}
__device__ __forceinline__ void async_wait() {
  asm volatile("s_wait_asynccnt 0" ::: "memory");
}

__device__ __forceinline__ float sp(float x) { return (x > 20.f) ? x : log1pf(expf(x)); }
__device__ __forceinline__ float gelu(float v) { return 0.5f * v * (1.f + erff(v * 0.70710678118f)); }

// ---------------------------------------------------------------------------
// Kernel 0: precompute complex multiplier, packed DFT bases, packed weights.
// blocks 0..63: multiplier; 64: BwP/BiP; 65: BhP/Einv; 66: packed weights
// ---------------------------------------------------------------------------
__global__ void k_init(const float* __restrict__ seg_h_h, const float* __restrict__ seg_h_w,
                       const float* __restrict__ nu_log, const float* __restrict__ alpha_log,
                       const float* __restrict__ c_log,  const float* __restrict__ omega_y,
                       const float* __restrict__ omega_x,
                       const float* __restrict__ amp_coef, const float* __restrict__ phase_coef,
                       const float* __restrict__ band_gain, const float* __restrict__ free_eps,
                       const float* __restrict__ fk_re, const float* __restrict__ fk_im,
                       const float* __restrict__ Wl, const float* __restrict__ W1,
                       const float* __restrict__ W2,
                       float* __restrict__ Mre, float* __restrict__ Mim,
                       h16* __restrict__ BwP, h16* __restrict__ BhP,
                       h16* __restrict__ Einv, h16* __restrict__ BiP,
                       h16* __restrict__ WlP, h16* __restrict__ W1P, h16* __restrict__ W2P) {
  const float TWO_PI = 6.28318530717958647692f;
  if (blockIdx.x < 64) {
    int c = blockIdx.x;
    float eh[4], ew[4], shh = 0.f, sww = 0.f;
#pragma unroll
    for (int i = 0; i < 4; ++i) { eh[i] = expf(seg_h_h[i]); shh += eh[i]; }
#pragma unroll
    for (int i = 0; i < 4; ++i) { ew[i] = expf(seg_h_w[i]); sww += ew[i]; }
    float gh[4], gw[4];
#pragma unroll
    for (int i = 0; i < 4; ++i) { gh[i] = eh[i] / shh * 4.f; gw[i] = ew[i] / sww * 4.f; }
    float nu0 = sp(nu_log[0]), alpha0 = sp(alpha_log[0]), c_amp = sp(c_log[0]);
    float g0 = sp(band_gain[0]), g1 = sp(band_gain[1]), g2 = sp(band_gain[2]);
    float oy = omega_y[0], ox = omega_x[0], eps = free_eps[0];
    float sa[4], pc[4];
#pragma unroll
    for (int k = 0; k < 4; ++k) { sa[k] = sp(amp_coef[c*4 + k]); pc[k] = phase_coef[c*4 + k]; }
    for (int t = threadIdx.x; t < 1024; t += 256) {
      int ky = t >> 5, kx = t & 31;
      float Ky = ky * (1.f / 31.f), Kx = kx * (1.f / 31.f);
      float r2 = Ky * Ky + Kx * Kx;
      float r  = sqrtf(r2 + 1e-12f);
      float amp_base = expf(-nu0 * r2) + c_amp / (1.f + alpha0 * r2 + 1e-6f);
      float w0v = (r <= 0.33f) ? 1.f : 0.f;
      float w2v = (r >= 0.66f) ? 1.f : 0.f;
      float w1v = fmaxf(1.f - w0v - w2v, 0.f);
      amp_base *= (1.f + g0) * w0v + (1.f + g1) * w1v + (1.f + g2) * w2v;
      float phi = oy * Ky + ox * Kx;
      float rk = 1.f, ad = 0.f, pd = 0.f;
#pragma unroll
      for (int k = 0; k < 4; ++k) { ad += sa[k] * rk; pd += pc[k] * rk; rk *= r; }
      float amp = amp_base * (1.f + fmaxf(ad, 0.f));
      phi += pd;
      float kr = cosf(phi) * amp, ki = sinf(phi) * amp;
      float gate = gh[ky >> 3] * gw[kx >> 3];
      float fr = fk_re[c*1024 + t], fi = fk_im[c*1024 + t];
      float ar = 1.f + eps * fr, ai = eps * fi;
      Mre[c*1024 + t] = gate * (kr * ar - ki * ai);
      Mim[c*1024 + t] = gate * (kr * ai + ki * ar);
    }
  } else if (blockIdx.x == 64) {
    // forward row basis, packed as B-tiles (k=w 256, n=64, NT=4)
    for (int t = threadIdx.x; t < 256*64; t += 256) {
      int w = t >> 6, j = t & 63;
      float v;
      if (j < 32) v =  cosf(TWO_PI * (float)(w * j) / 256.f);
      else        v = -sinf(TWO_PI * (float)(w * (j - 32)) / 256.f);
      BwP[packB(w, j, 4)] = (h16)v;
    }
    // inverse row basis, packed as B-tiles (k=64, n=w 256, NT=16), scaled s_k/W
    for (int t = threadIdx.x; t < 64*256; t += 256) {
      int j = t >> 8, w = t & 255;
      int f = (j < 32) ? j : (j - 32);
      float s = (f == 0) ? (1.f / 256.f) : (2.f / 256.f);
      float ang = TWO_PI * (float)(w * f) / 256.f;
      float v = (j < 32) ? s * cosf(ang) : -s * sinf(ang);
      BiP[packB(j, w, 16)] = (h16)v;
    }
  } else if (blockIdx.x == 65) {
    // forward column basis, packed as A-tiles (m=32 coeff rows, k=h 256, KT=8)
    for (int t = threadIdx.x; t < 32*256; t += 256) {
      int row = t >> 8, hh = t & 255;
      int f = (row < 16) ? row : (240 + (row - 16));
      float ang = TWO_PI * (float)(hh * f) / 256.f;
      int po = packA(row, hh, 8);
      BhP[po]        = (h16)cosf(ang);
      BhP[8192 + po] = (h16)(-sinf(ang));
    }
    // inverse column basis: row-major [h][64] (re|im), scaled 1/H
    for (int t = threadIdx.x; t < 256*32; t += 256) {
      int hh = t >> 5, j = t & 31;
      int f = (j < 16) ? j : (240 + (j - 16));
      float ang = TWO_PI * (float)(hh * f) / 256.f;
      Einv[hh*64 + j]      = (h16)(cosf(ang) * (1.f / 256.f));
      Einv[hh*64 + 32 + j] = (h16)(sinf(ang) * (1.f / 256.f));
    }
  } else {
    // packed weights: B[k][n] = W[n][k]
    for (int t = threadIdx.x; t < 4096; t += 256) {
      int c = t >> 6, o = t & 63;
      WlP[packB(c, o, 4)] = (h16)Wl[o*64 + c];
    }
    for (int t = threadIdx.x; t < 8192; t += 256) {
      int c = t >> 7, o = t & 127;
      W1P[packB(c, o, 8)] = (h16)W1[o*64 + c];
    }
    for (int t = threadIdx.x; t < 8192; t += 256) {
      int k = t >> 6, o = t & 63;
      W2P[packB(k, o, 4)] = (h16)W2[o*128 + k];
    }
  }
}

// ---------------------------------------------------------------------------
// Kernel 1: row DFT. GEMM M=65536, K=256 (w), N=64. X1 written packed (NT=4).
// ---------------------------------------------------------------------------
__global__ void k_rowdft(const float* __restrict__ x, const h16* __restrict__ BwP,
                         h16* __restrict__ X1) {
  __shared__ alignas(16) h16 As[64 * 256];
  __shared__ alignas(16) h16 Bs[256 * 64];
  const int tid = threadIdx.x, lane = tid & 31, wv = tid >> 5;
  const long base = (long)blockIdx.x * 64 * 256;
  async_issue_f16(Bs, BwP, 256 * 64, tid, 128);
  __builtin_prefetch(&x[base + 64 * 256], 0, 1);
  for (int i8 = tid; i8 < (64 * 256) / 8; i8 += 128) {
    const float* g = &x[base + (i8 << 3)];
    v8h v;
#pragma unroll
    for (int j = 0; j < 8; ++j) v[j] = (h16)g[j];
    *(v8h*)&As[i8 << 3] = v;
  }
  async_wait();
  __syncthreads();
  v8f acc[4];
#pragma unroll
  for (int nt = 0; nt < 4; ++nt) acc[nt] = vzero8();
  for (int kt = 0; kt < 8; ++kt) {
    v16h a = frag_A(&As[wv * 16 * 256 + kt * 32], 256, lane);
#pragma unroll
    for (int nt = 0; nt < 4; ++nt) {
      v16h b = frag_P(Bs, kt * 4 + nt, lane);
      acc[nt] = WMMA(a, b, acc[nt]);
    }
  }
  const int mb = (lane >> 4) << 3, n = lane & 15;
#pragma unroll
  for (int nt = 0; nt < 4; ++nt)
#pragma unroll
    for (int r = 0; r < 8; ++r) {
      int row = (int)blockIdx.x * 64 + wv * 16 + mb + r;
      int bc = row >> 8, hh = row & 255;
      X1[(long)bc * 16384 + packB(hh, nt * 16 + n, 4)] = (h16)acc[nt][r];
    }
}

// ---------------------------------------------------------------------------
// Kernel 2: column DFT (complex) + spectral multiplier. One block per (b,c).
// ---------------------------------------------------------------------------
__global__ void k_coldft(const h16* __restrict__ X1, const h16* __restrict__ BhP,
                         const float* __restrict__ Mre, const float* __restrict__ Mim,
                         float* __restrict__ Cc) {
  __shared__ alignas(16) h16 Xs[256 * 64];
  __shared__ alignas(16) h16 Bs[2 * 32 * 256];
  const int tid = threadIdx.x, lane = tid & 31, wv = tid >> 5;
  const int bc = blockIdx.x;
  async_issue_f16(Xs, X1 + (long)bc * 16384, 16384, tid, 128);
  async_issue_f16(Bs, BhP, 16384, tid, 128);
  async_wait();
  __syncthreads();
  const int mi = wv >> 1, ni = wv & 1;
  v8f p1 = vzero8(), p2 = vzero8(), p3 = vzero8(), p4 = vzero8();
  for (int kt = 0; kt < 8; ++kt) {
    v16h aR = frag_P(Bs, mi * 8 + kt, lane);
    v16h aI = frag_P(Bs + 8192, mi * 8 + kt, lane);
    v16h bR = frag_P(Xs, kt * 4 + ni, lane);
    v16h bI = frag_P(Xs, kt * 4 + 2 + ni, lane);
    p1 = WMMA(aR, bR, p1);
    p2 = WMMA(aI, bI, p2);
    p3 = WMMA(aR, bI, p3);
    p4 = WMMA(aI, bR, p4);
  }
  const int c = bc & 63;
  const int mb = (lane >> 4) << 3, n = lane & 15;
#pragma unroll
  for (int r = 0; r < 8; ++r) {
    int ky = mi * 16 + mb + r;
    int kx = ni * 16 + n;
    float cr = p1[r] - p2[r];
    float ci = p3[r] + p4[r];
    float mr = Mre[c * 1024 + ky * 32 + kx];
    float mm = Mim[c * 1024 + ky * 32 + kx];
    Cc[(long)bc * 2048 + ky * 32 + kx]        = cr * mr - ci * mm;
    Cc[(long)bc * 2048 + 1024 + ky * 32 + kx] = cr * mm + ci * mr;
  }
}

// ---------------------------------------------------------------------------
// Kernel 3: inverse column DFT. One block per (b,c). Y1 row-major [h][64].
// ---------------------------------------------------------------------------
__global__ void k_icoldft(const float* __restrict__ Cc, const h16* __restrict__ Einv,
                          h16* __restrict__ Y1) {
  __shared__ alignas(16) h16 Es[256 * 64];
  __shared__ alignas(16) h16 Cs[4 * 512];    // packed B tiles: re0,re1,im0,im1
  const int tid = threadIdx.x, lane = tid & 31, wv = tid >> 5;
  const int bc = blockIdx.x;
  async_issue_f16(Es, Einv, 16384, tid, 128);
  for (int i = tid; i < 1024; i += 128) {
    int ky = i >> 5, kx = i & 31;
    Cs[packB(ky, kx, 2)]      = (h16)Cc[(long)bc * 2048 + i];
    Cs[packB(ky, kx + 32, 2)] = (h16)Cc[(long)bc * 2048 + 1024 + i];
  }
  async_wait();
  __syncthreads();
  const int mb = (lane >> 4) << 3, n = lane & 15;
  for (int t = 0; t < 4; ++t) {
    int mt = wv + t * 4;
    v16h aR = frag_A(&Es[mt * 16 * 64 + 0], 64, lane);
    v16h aI = frag_A(&Es[mt * 16 * 64 + 32], 64, lane);
#pragma unroll
    for (int nt = 0; nt < 2; ++nt) {
      v16h bR = frag_P(Cs, nt, lane);
      v16h bI = frag_P(Cs, 2 + nt, lane);
      v8f p1 = WMMA(aR, bR, vzero8());
      v8f p2 = WMMA(aI, bI, vzero8());
      v8f p3 = WMMA(aR, bI, vzero8());
      v8f p4 = WMMA(aI, bR, vzero8());
#pragma unroll
      for (int r = 0; r < 8; ++r) {
        long row = (long)bc * 256 + mt * 16 + mb + r;
        Y1[row * 64 + nt * 16 + n]      = (h16)(p1[r] - p2[r]);
        Y1[row * 64 + 32 + nt * 16 + n] = (h16)(p3[r] + p4[r]);
      }
    }
  }
}

// ---------------------------------------------------------------------------
// Kernel 4: inverse row DFT. GEMM M=65536, K=64, N=256. Writes y into Z.
// ---------------------------------------------------------------------------
__global__ void k_irowdft(const h16* __restrict__ Y1, const h16* __restrict__ BiP,
                          float* __restrict__ Z) {
  __shared__ alignas(16) h16 As[64 * 64];
  __shared__ alignas(16) h16 Bs[64 * 256];
  const int tid = threadIdx.x, lane = tid & 31, wv = tid >> 5;
  const long m0 = (long)blockIdx.x * 64;
  async_issue_f16(As, Y1 + m0 * 64, 4096, tid, 128);
  async_issue_f16(Bs, BiP, 16384, tid, 128);
  async_wait();
  __syncthreads();
  const int mb = (lane >> 4) << 3, n = lane & 15;
  v16h a0 = frag_A(&As[wv * 16 * 64 + 0], 64, lane);
  v16h a1 = frag_A(&As[wv * 16 * 64 + 32], 64, lane);
  for (int p = 0; p < 2; ++p) {
    v8f acc[8];
#pragma unroll
    for (int j = 0; j < 8; ++j) acc[j] = vzero8();
#pragma unroll
    for (int j = 0; j < 8; ++j) {
      int nt = p * 8 + j;
      acc[j] = WMMA(a0, frag_P(Bs, nt, lane), acc[j]);
      acc[j] = WMMA(a1, frag_P(Bs, 16 + nt, lane), acc[j]);
    }
#pragma unroll
    for (int j = 0; j < 8; ++j)
#pragma unroll
      for (int r = 0; r < 8; ++r) {
        long row = m0 + wv * 16 + mb + r;
        Z[row * 256 + (p * 8 + j) * 16 + n] = acc[j][r];
      }
  }
}

// ---------------------------------------------------------------------------
// Kernel 5: fused channel-mix (1x1 local, MLP 64->128->64, depthwise 3x3),
// accumulating into Z (holds y). One block per (b, h, 64-px tile).
// ---------------------------------------------------------------------------
__global__ void k_mix(const float* __restrict__ x,
                      const h16* __restrict__ WlP, const float* __restrict__ bl,
                      const float* __restrict__ Wd, const float* __restrict__ bd,
                      const h16* __restrict__ W1P, const float* __restrict__ b1,
                      const h16* __restrict__ W2P, const float* __restrict__ b2,
                      float* __restrict__ Z) {
  __shared__ alignas(16) h16 Xp[64 * 64];     // [px][c]
  __shared__ alignas(16) h16 H1[64 * 128];    // [px][hidden]
  __shared__ alignas(16) h16 Wp[20480];       // WlP(4096) | W1P(8192) | W2P(8192)
  const int tid = threadIdx.x, lane = tid & 31, wv = tid >> 5;
  const int wt = blockIdx.x & 3;
  const int h  = (blockIdx.x >> 2) & 255;
  const int b  = blockIdx.x >> 10;
  const int w0 = wt * 64;
  async_issue_f16(Wp, WlP, 4096, tid, 128);
  async_issue_f16(Wp + 4096, W1P, 8192, tid, 128);
  async_issue_f16(Wp + 12288, W2P, 8192, tid, 128);
  for (int i = tid; i < 64 * 64; i += 128) {
    int c = i >> 6, px = i & 63;
    Xp[px * 64 + c] = (h16)x[((long)(b * 64 + c) * 256 + h) * 256 + w0 + px];
  }
  async_wait();
  __syncthreads();
  const int mb = (lane >> 4) << 3, n16 = lane & 15;
  v16h ax0 = frag_A(&Xp[wv * 16 * 64 + 0], 64, lane);
  v16h ax1 = frag_A(&Xp[wv * 16 * 64 + 32], 64, lane);
  // local = x @ Wl^T
  v8f locf[4];
#pragma unroll
  for (int nt = 0; nt < 4; ++nt) {
    v8f acc = vzero8();
    acc = WMMA(ax0, frag_P(Wp, nt, lane), acc);
    acc = WMMA(ax1, frag_P(Wp, 4 + nt, lane), acc);
    locf[nt] = acc;
  }
  // h1 = gelu(x @ W1^T + b1) -> LDS f16
#pragma unroll
  for (int nt = 0; nt < 8; ++nt) {
    v8f acc = vzero8();
    acc = WMMA(ax0, frag_P(Wp + 4096, nt, lane), acc);
    acc = WMMA(ax1, frag_P(Wp + 4096, 8 + nt, lane), acc);
    int col = nt * 16 + n16;
    float bias = b1[col];
#pragma unroll
    for (int r = 0; r < 8; ++r) {
      float v = gelu(acc[r] + bias);
      H1[(wv * 16 + mb + r) * 128 + col] = (h16)v;
    }
  }
  __syncthreads();
  // cmlp = h1 @ W2^T
  v8f cf[4];
#pragma unroll
  for (int nt = 0; nt < 4; ++nt) {
    v8f acc = vzero8();
#pragma unroll
    for (int kt = 0; kt < 4; ++kt) {
      v16h a = frag_A(&H1[wv * 16 * 128 + kt * 32], 128, lane);
      acc = WMMA(a, frag_P(Wp + 12288, kt * 4 + nt, lane), acc);
    }
    cf[nt] = acc;
  }
  // combine: z = y + 0.3*(local+bl) + 0.15*dw + (cmlp+b2)
#pragma unroll
  for (int nt = 0; nt < 4; ++nt) {
    int o = nt * 16 + n16;
    float blo = bl[o], b2o = b2[o], bdo = bd[o];
    float wd9[9];
#pragma unroll
    for (int i = 0; i < 9; ++i) wd9[i] = Wd[o * 9 + i];
    const float* xo = x + (long)(b * 64 + o) * 256 * 256;
#pragma unroll
    for (int r = 0; r < 8; ++r) {
      int px = wv * 16 + mb + r;
      int ww = w0 + px;
      float dwv = bdo;
#pragma unroll
      for (int dy = 0; dy < 3; ++dy) {
        int hh = h + dy - 1;
        if ((unsigned)hh < 256u) {
#pragma unroll
          for (int dx = 0; dx < 3; ++dx) {
            int wx = ww + dx - 1;
            if ((unsigned)wx < 256u) dwv += wd9[dy * 3 + dx] * xo[hh * 256 + wx];
          }
        }
      }
      long za = ((long)(b * 64 + o) * 256 + h) * 256 + ww;
      float z = Z[za] + 0.3f * (locf[nt][r] + blo) + 0.15f * dwv + (cf[nt][r] + b2o);
      Z[za] = z;
    }
  }
}

// ---------------------------------------------------------------------------
// Kernel 6: instance norm + gelu + residual. One block per (b,c).
// ---------------------------------------------------------------------------
__global__ void k_finalize(const float* __restrict__ x, const float* __restrict__ Z,
                           const float* __restrict__ gamma, const float* __restrict__ beta,
                           float* __restrict__ out) {
  __shared__ float s1[256], s2[256];
  const int bc = blockIdx.x, tid = threadIdx.x;
  const long base = (long)bc * 65536;
  float a = 0.f, bb = 0.f;
  for (int i = tid; i < 65536; i += 256) {
    float v = Z[base + i];
    a += v; bb += v * v;
  }
  s1[tid] = a; s2[tid] = bb;
  __syncthreads();
  for (int s = 128; s > 0; s >>= 1) {
    if (tid < s) { s1[tid] += s1[tid + s]; s2[tid] += s2[tid + s]; }
    __syncthreads();
  }
  float mu  = s1[0] * (1.f / 65536.f);
  float var = s2[0] * (1.f / 65536.f) - mu * mu;
  float rs  = rsqrtf(var + 1e-5f);
  int c = bc & 63;
  float g = gamma[c] * rs, be = beta[c];
  for (int i = tid; i < 65536; i += 256) {
    float zn = (Z[base + i] - mu) * g + be;
    out[base + i] = x[base + i] + gelu(zn);
  }
}

// ---------------------------------------------------------------------------
extern "C" void kernel_launch(void* const* d_in, const int* in_sizes, int n_in,
                              void* d_out, int out_size, void* d_ws, size_t ws_size,
                              hipStream_t stream) {
  (void)in_sizes; (void)n_in; (void)out_size; (void)ws_size;
  const float* x         = (const float*)d_in[0];
  const float* seg_h_h   = (const float*)d_in[1];
  const float* seg_h_w   = (const float*)d_in[2];
  const float* nu_log    = (const float*)d_in[3];
  const float* alpha_log = (const float*)d_in[4];
  const float* c_log     = (const float*)d_in[5];
  const float* omega_y   = (const float*)d_in[6];
  const float* omega_x   = (const float*)d_in[7];
  const float* amp_coef  = (const float*)d_in[8];
  const float* phase_coef= (const float*)d_in[9];
  const float* band_gain = (const float*)d_in[10];
  const float* free_eps  = (const float*)d_in[11];
  const float* fk_re     = (const float*)d_in[12];
  const float* fk_im     = (const float*)d_in[13];
  const float* w_local   = (const float*)d_in[14];
  const float* b_local   = (const float*)d_in[15];
  const float* w_dw3     = (const float*)d_in[16];
  const float* b_dw3     = (const float*)d_in[17];
  const float* w_mlp1    = (const float*)d_in[18];
  const float* b_mlp1    = (const float*)d_in[19];
  const float* w_mlp2    = (const float*)d_in[20];
  const float* b_mlp2    = (const float*)d_in[21];
  const float* gamma     = (const float*)d_in[22];
  const float* beta      = (const float*)d_in[23];
  float* out = (float*)d_out;

  char* ws = (char*)d_ws;
  size_t off = 0;
  auto alloc = [&](size_t bytes) -> char* {
    off = (off + 255) & ~(size_t)255;
    char* p = ws + off;
    off += bytes;
    return p;
  };
  float* Mre  = (float*)alloc(64 * 1024 * sizeof(float));
  float* Mim  = (float*)alloc(64 * 1024 * sizeof(float));
  h16*   BwP  = (h16*)alloc(256 * 64 * sizeof(h16));
  h16*   BhP  = (h16*)alloc(2 * 32 * 256 * sizeof(h16));
  h16*   Einv = (h16*)alloc(256 * 64 * sizeof(h16));
  h16*   BiP  = (h16*)alloc(64 * 256 * sizeof(h16));
  h16*   WlP  = (h16*)alloc(4096 * sizeof(h16));
  h16*   W1P  = (h16*)alloc(8192 * sizeof(h16));
  h16*   W2P  = (h16*)alloc(8192 * sizeof(h16));
  h16*   X1   = (h16*)alloc((size_t)65536 * 64 * sizeof(h16));   // reused as Y1
  float* Cc   = (float*)alloc((size_t)256 * 2048 * sizeof(float));
  float* Z    = (float*)alloc((size_t)4 * 64 * 256 * 256 * sizeof(float));

  k_init<<<67, 256, 0, stream>>>(seg_h_h, seg_h_w, nu_log, alpha_log, c_log,
                                 omega_y, omega_x, amp_coef, phase_coef, band_gain,
                                 free_eps, fk_re, fk_im, w_local, w_mlp1, w_mlp2,
                                 Mre, Mim, BwP, BhP, Einv, BiP, WlP, W1P, W2P);
  k_rowdft<<<1024, 128, 0, stream>>>(x, BwP, X1);
  k_coldft<<<256, 128, 0, stream>>>(X1, BhP, Mre, Mim, Cc);
  k_icoldft<<<256, 128, 0, stream>>>(Cc, Einv, X1 /* reused as Y1 */);
  k_irowdft<<<1024, 128, 0, stream>>>(X1, BiP, Z);
  k_mix<<<4096, 128, 0, stream>>>(x, WlP, b_local, w_dw3, b_dw3,
                                  W1P, b_mlp1, W2P, b_mlp2, Z);
  k_finalize<<<256, 256, 0, stream>>>(x, Z, gamma, beta, out);
}
